// SAGE_9878424780960
// MI455X (gfx1250) — matrix operand order repaired and checked by reading, not compiled, not run on hardware
//
#include <hip/hip_runtime.h>
#include <hip/hip_bf16.h>
#include <math.h>

// ---------------------------------------------------------------------------
// GraphSAGE 3-layer forward for MI455X (gfx1250, wave32, WMMA).
// Memory-bound: layer-0 edge gather/scatter (~1.3 GB) dominates at 23.3 TB/s.
// GEMMs (~19 GFLOP total) run on v_wmma_f32_16x16x32_bf16 with f32 accum;
// W tiles staged in LDS (bf16) once per block and shared by all 8 waves.
// ---------------------------------------------------------------------------

typedef __bf16 bf16x16 __attribute__((ext_vector_type(16)));
typedef __bf16 bf16x2  __attribute__((ext_vector_type(2)));
typedef float  f32x8   __attribute__((ext_vector_type(8)));

static constexpr int kN1 = 123904;
static constexpr int kN2 = 11264;
static constexpr int kN3 = 1024;
static constexpr int kInC  = 128;
static constexpr int kHidC = 256;
static constexpr int kOutC = 47;

static constexpr int KMAX = 256;   // max K over all layers
static constexpr int BPAD = 16;    // bf16 pad per LDS row: keeps 32B alignment + bank spread

// ---------------- scatter: agg[tgt] += x[src] (float4 per thread) ----------
__global__ __launch_bounds__(256)
void sage_scatter_add4(const float* __restrict__ x, const int* __restrict__ ei,
                       int E, int c4shift, float* __restrict__ agg) {
    int tid = blockIdx.x * blockDim.x + threadIdx.x;
    int total = E << c4shift;                 // E * (C/4)
    if (tid >= total) return;
    int e  = tid >> c4shift;
    int c4 = tid & ((1 << c4shift) - 1);
    int src = ei[e];
    int tgt = ei[E + e];
    int C = 4 << c4shift;
    const float4 v = *(const float4*)(x + src * C + 4 * c4);     // int offsets (< 2^31)
    float* dst = agg + tgt * C + 4 * c4;
    atomicAdd(dst + 0, v.x);
    atomicAdd(dst + 1, v.y);
    atomicAdd(dst + 2, v.z);
    atomicAdd(dst + 3, v.w);
}

__global__ __launch_bounds__(256)
void sage_scatter_count(const int* __restrict__ ei, int E, float* __restrict__ cnt) {
    int e = blockIdx.x * blockDim.x + threadIdx.x;
    if (e < E) atomicAdd(&cnt[ei[E + e]], 1.0f);
}

// agg[row][c] /= max(cnt[row], 1)
__global__ __launch_bounds__(256)
void sage_normalize(float* __restrict__ agg, const float* __restrict__ cnt,
                    int rows, int cshift) {
    int tid = blockIdx.x * blockDim.x + threadIdx.x;
    int total = rows << cshift;
    if (tid >= total) return;
    int row = tid >> cshift;
    agg[tid] *= 1.0f / fmaxf(cnt[row], 1.0f);
}

// ---------------- WMMA GEMM: C[M,N] (+)= A[M,K] @ W[K,N] (+ bias) ----------
// flags bit0: accumulate into existing C; bit1: ReLU on store.
// Block: 256 threads = 8 waves; each wave computes one 16x16 C tile; block
// covers 128 rows x 16 cols. M must be a multiple of 128 (123904/11264/1024).
// W[:, n0:n0+16] is staged once per block into LDS as bf16 in [n][k] layout.
__global__ __launch_bounds__(256)
void sage_wmma_gemm(const float* __restrict__ A, const float* __restrict__ W,
                    const float* __restrict__ bias, float* __restrict__ C,
                    int M, int K, int Ndim, int flags) {
    __shared__ alignas(64) __bf16 Blds[16 * (KMAX + BPAD)];

    const int tid = threadIdx.x;
    const int n0  = blockIdx.y * 16;

    // ---- stage W tile -> LDS (bf16), coalesced reads, packed b32 stores ----
    {
        const int n     = tid & 15;                 // column within the 16-wide tile
        const int col   = n0 + n;
        const bool cok  = col < Ndim;
        const int ccol  = cok ? col : (Ndim - 1);   // clamp address, zero value
        __bf16* dst = Blds + n * (KMAX + BPAD);
        for (int k = (tid >> 4) * 2; k < K; k += 32) {
            float w0 = W[k * Ndim + ccol];
            float w1 = W[(k + 1) * Ndim + ccol];
            if (!cok) { w0 = 0.0f; w1 = 0.0f; }
            bf16x2 p;
            p[0] = (__bf16)w0;
            p[1] = (__bf16)w1;
            *(bf16x2*)(dst + k) = p;                // 4B-aligned (k even, row stride 544B)
        }
    }
    __syncthreads();

    const int wave = tid >> 5;
    const int lane = tid & 31;
    const int half = lane >> 4;       // which 16-lane group
    const int mn   = lane & 15;       // m index for A, n index for B/C
    const int m0   = (blockIdx.x * 8 + wave) * 16;   // always < M (grid exact)

    const int  col   = n0 + mn;
    const bool colok = col < Ndim;

    // init accumulator (C layout: acc[v] = C[m0 + v + 8*half][n0 + mn])
    f32x8 acc;
    if (flags & 1) {
#pragma unroll
        for (int v = 0; v < 8; ++v) {
            int row = m0 + v + 8 * half;
            acc[v] = colok ? C[row * Ndim + col] : 0.0f;
        }
    } else {
        float bv = colok ? bias[col] : 0.0f;
#pragma unroll
        for (int v = 0; v < 8; ++v) acc[v] = bv;
    }

    const float*  Arow = A + (m0 + mn) * K;              // int offsets (< 2^31)
    const __bf16* Brow = Blds + mn * (KMAX + BPAD) + 16 * half;

    for (int k0 = 0; k0 < K; k0 += 32) {
        // WGP-scope prefetch of the next A chunk (global_prefetch_b8, SCOPE_WGP)
        if (k0 + 32 < K) __builtin_prefetch(Arow + k0 + 32, 0, 3);
        // A fragment (16x32 bf16): lane holds row m = mn.
        //   elems 0..7  -> K = k0 + 16*half + e       (VGPRs 0-3)
        //   elems 8..15 -> K = k0 + 16*half + 8 + e   (VGPRs 4-7)
        bf16x16 a;
#pragma unroll
        for (int e = 0; e < 8; ++e) {
            a[e]     = (__bf16)Arow[k0 + 16 * half + e];
            a[8 + e] = (__bf16)Arow[k0 + 16 * half + 8 + e];
        }
        // B fragment (32x16 bf16): lane holds column n = mn,
        //   elems 0..15 -> K = k0 + 16*half + e  => contiguous 32B in LDS.
        bf16x16 b = *(const bf16x16*)(Brow + k0);        // 32B aligned
        acc = __builtin_amdgcn_wmma_f32_16x16x32_bf16(
            /*neg_a=*/false, a, /*neg_b=*/false, b,
            /*c_mod=*/(short)0, acc, /*reuse_a=*/false, /*reuse_b=*/false);
    }

#pragma unroll
    for (int v = 0; v < 8; ++v) {
        int row = m0 + v + 8 * half;
        float r = acc[v];
        if (flags & 2) r = fmaxf(r, 0.0f);
        if (colok) C[row * Ndim + col] = r;
    }
}

// ---------------- log_softmax over 47 columns, one thread per row ----------
__global__ __launch_bounds__(256)
void sage_log_softmax(const float* __restrict__ logits, float* __restrict__ out,
                      int rows, int C) {
    int r = blockIdx.x * blockDim.x + threadIdx.x;
    if (r >= rows) return;
    const float* p = logits + r * C;
    float mx = -INFINITY;
    for (int c = 0; c < C; ++c) mx = fmaxf(mx, p[c]);
    float s = 0.0f;
    for (int c = 0; c < C; ++c) s += __expf(p[c] - mx);
    float lse = __logf(s) + mx;
    float* o = out + r * C;
    for (int c = 0; c < C; ++c) o[c] = p[c] - lse;
}

// ---------------------------------------------------------------------------
static inline size_t alignup(size_t v) { return (v + 255) & ~(size_t)255; }

extern "C" void kernel_launch(void* const* d_in, const int* in_sizes, int n_in,
                              void* d_out, int out_size, void* d_ws, size_t ws_size,
                              hipStream_t stream) {
    const float* x   = (const float*)d_in[0];
    const int*   ei0 = (const int*)d_in[1];
    const int*   ei1 = (const int*)d_in[2];
    const int*   ei2 = (const int*)d_in[3];
    const float* Wl0 = (const float*)d_in[4];
    const float* Wr0 = (const float*)d_in[5];
    const float* b0  = (const float*)d_in[6];
    const float* Wl1 = (const float*)d_in[7];
    const float* Wr1 = (const float*)d_in[8];
    const float* b1  = (const float*)d_in[9];
    const float* Wl2 = (const float*)d_in[10];
    const float* Wr2 = (const float*)d_in[11];
    const float* b2  = (const float*)d_in[12];

    const int E0 = in_sizes[1] / 2;   // 1239040
    const int E1 = in_sizes[2] / 2;   // 112640
    const int E2 = in_sizes[3] / 2;   // 10240

    // workspace carve-out
    char* ws = (char*)d_ws;
    size_t off = 0;
    auto take = [&](size_t bytes) { char* p = ws + off; off = alignup(off + bytes); return p; };
    float* agg0   = (float*)take((size_t)kN1 * kInC  * 4);
    float* cnt0   = (float*)take((size_t)kN1 * 4);
    float* h1     = (float*)take((size_t)kN1 * kHidC * 4);
    float* agg1   = (float*)take((size_t)kN2 * kHidC * 4);
    float* cnt1   = (float*)take((size_t)kN2 * 4);
    float* h2     = (float*)take((size_t)kN2 * kHidC * 4);
    float* agg2   = (float*)take((size_t)kN3 * kHidC * 4);
    float* cnt2   = (float*)take((size_t)kN3 * 4);
    float* logits = (float*)take((size_t)kN3 * kOutC * 4);
    (void)ws_size;

    // zero accumulation buffers every call (graph-capture safe)
    hipMemsetAsync(agg0, 0, (size_t)kN1 * kInC  * 4, stream);
    hipMemsetAsync(cnt0, 0, (size_t)kN1 * 4, stream);
    hipMemsetAsync(agg1, 0, (size_t)kN2 * kHidC * 4, stream);
    hipMemsetAsync(cnt1, 0, (size_t)kN2 * 4, stream);
    hipMemsetAsync(agg2, 0, (size_t)kN3 * kHidC * 4, stream);
    hipMemsetAsync(cnt2, 0, (size_t)kN3 * 4, stream);

    const int TB = 256;
    auto blocks = [](long long n, int tb) { return (unsigned)((n + tb - 1) / tb); };

    // ---- layer 0: x(N0x128) --ei0--> agg0(N1x128); h1 = relu(agg0@Wl0 + x@Wr0 + b0)
    sage_scatter_add4<<<blocks((long long)E0 * (kInC / 4), TB), TB, 0, stream>>>(
        x, ei0, E0, /*c4shift=*/5, agg0);
    sage_scatter_count<<<blocks(E0, TB), TB, 0, stream>>>(ei0, E0, cnt0);
    sage_normalize<<<blocks((long long)kN1 * kInC, TB), TB, 0, stream>>>(agg0, cnt0, kN1, 7);
    {
        dim3 g(kN1 / 128, kHidC / 16);
        sage_wmma_gemm<<<g, TB, 0, stream>>>(agg0, Wl0, b0, h1, kN1, kInC, kHidC, 0);
        sage_wmma_gemm<<<g, TB, 0, stream>>>(x,    Wr0, nullptr, h1, kN1, kInC, kHidC, 1 | 2);
    }

    // ---- layer 1: h1(N1x256) --ei1--> agg1(N2x256); h2 = relu(agg1@Wl1 + h1@Wr1 + b1)
    sage_scatter_add4<<<blocks((long long)E1 * (kHidC / 4), TB), TB, 0, stream>>>(
        h1, ei1, E1, /*c4shift=*/6, agg1);
    sage_scatter_count<<<blocks(E1, TB), TB, 0, stream>>>(ei1, E1, cnt1);
    sage_normalize<<<blocks((long long)kN2 * kHidC, TB), TB, 0, stream>>>(agg1, cnt1, kN2, 8);
    {
        dim3 g(kN2 / 128, kHidC / 16);
        sage_wmma_gemm<<<g, TB, 0, stream>>>(agg1, Wl1, b1, h2, kN2, kHidC, kHidC, 0);
        sage_wmma_gemm<<<g, TB, 0, stream>>>(h1,   Wr1, nullptr, h2, kN2, kHidC, kHidC, 1 | 2);
    }

    // ---- layer 2: h2(N2x256) --ei2--> agg2(N3x256); logits = agg2@Wl2 + h2@Wr2 + b2
    sage_scatter_add4<<<blocks((long long)E2 * (kHidC / 4), TB), TB, 0, stream>>>(
        h2, ei2, E2, /*c4shift=*/6, agg2);
    sage_scatter_count<<<blocks(E2, TB), TB, 0, stream>>>(ei2, E2, cnt2);
    sage_normalize<<<blocks((long long)kN3 * kHidC, TB), TB, 0, stream>>>(agg2, cnt2, kN3, 8);
    {
        dim3 g(kN3 / 128, (kOutC + 15) / 16);   // N=47 -> 3 column tiles, guarded
        sage_wmma_gemm<<<g, TB, 0, stream>>>(agg2, Wl2, b2, logits, kN3, kHidC, kOutC, 0);
        sage_wmma_gemm<<<g, TB, 0, stream>>>(h2,   Wr2, nullptr, logits, kN3, kHidC, kOutC, 1);
    }

    // ---- log_softmax over classes -> d_out
    sage_log_softmax<<<blocks(kN3, TB), TB, 0, stream>>>(logits, (float*)d_out, kN3, kOutC);
}
